// AttentionBlock_56813827392062
// MI455X (gfx1250) — matrix operand order repaired and checked by reading, not compiled, not run on hardware
//
#include <hip/hip_runtime.h>
#include <hip/hip_bf16.h>
#include <stdint.h>

// ---------------------------------------------------------------------------
// Problem constants (match reference)
// ---------------------------------------------------------------------------
#define BB 4
#define TT 2048
#define DD 1024
#define HH 16
#define HD 64

typedef __attribute__((ext_vector_type(16))) __bf16 v16bf;
typedef __attribute__((ext_vector_type(8)))  float  v8f;
typedef __attribute__((ext_vector_type(4)))  float  v4f;
typedef __attribute__((ext_vector_type(4)))  unsigned int v4u;
typedef __attribute__((ext_vector_type(2)))  unsigned int v2u;

union FragBF16 {
    v16bf v;
    v4u   q[2];
    unsigned int u[8];
};

__device__ __forceinline__ unsigned int f2bf_u32(float x) {
    unsigned int u = __float_as_uint(x);
    unsigned int r = u + 0x7FFFu + ((u >> 16) & 1u);
    return r >> 16;
}
__device__ __forceinline__ unsigned short f2bf(float x) {
    return (unsigned short)f2bf_u32(x);
}
__device__ __forceinline__ unsigned int pack_bf16(float lo, float hi) {
    return f2bf_u32(lo) | (f2bf_u32(hi) << 16);
}

// A-fragment (16x32 bf16, M x K). lane L: row = L&15, K chunks of 8 at
// kbase=(L>>4)*8 and kbase+16.  (ISA 7.12.2 "16-bit A-Matrix 16x32")
__device__ __forceinline__ void load_frag_a(FragBF16& f,
                                            const unsigned short* base,
                                            int row_stride, int lane) {
    const int row = lane & 15;
    const int kb  = (lane >> 4) * 8;
    f.q[0] = *reinterpret_cast<const v4u*>(base + (size_t)row * row_stride + kb);
    f.q[1] = *reinterpret_cast<const v4u*>(base + (size_t)row * row_stride + kb + 16);
}

// B-fragment (32x16 bf16, K x N) from an N-major (row = N, contiguous K)
// matrix: lane L: col = L&15, K chunks of 8 at kbase=(L>>4)*8 and kbase+16.
__device__ __forceinline__ void load_frag_b(FragBF16& f,
                                            const unsigned short* base,
                                            int row_stride, int lane) {
    const int col = lane & 15;
    const int kb  = (lane >> 4) * 8;
    f.q[0] = *reinterpret_cast<const v4u*>(base + (size_t)col * row_stride + kb);
    f.q[1] = *reinterpret_cast<const v4u*>(base + (size_t)col * row_stride + kb + 16);
}

__device__ __forceinline__ v8f wmma_bf16(const FragBF16& a, const FragBF16& b, v8f c) {
    return __builtin_amdgcn_wmma_f32_16x16x32_bf16(
        false, a.v, false, b.v, (short)0, c, false, false);
}

// ---------------------------------------------------------------------------
// Kernel 1: fp32 -> bf16 conversion, 4 elements/thread
// ---------------------------------------------------------------------------
__global__ void __launch_bounds__(256)
cvt_f32_bf16x4(const float* __restrict__ in,
               unsigned short* __restrict__ out, int n4) {
    int i = blockIdx.x * blockDim.x + threadIdx.x;
    if (i < n4) {
        v4f a = reinterpret_cast<const v4f*>(in)[i];
        v2u o;
        o[0] = pack_bf16(a[0], a[1]);
        o[1] = pack_bf16(a[2], a[3]);
        reinterpret_cast<v2u*>(out)[i] = o;
    }
}

// ---------------------------------------------------------------------------
// Kernel 2: NT GEMM  C[m,n] = sum_k A[m,k]*Bw[n,k] + bias[n]
//   64x64 tile per wave, K-step 32 -> 16 WMMAs per step, 16 b128 loads.
//   MODE 0: bf16 out scattered to [B,H,T,HD]   (Q, K projections)
//   MODE 1: bf16 out scattered to [B,H,HD,T]   (V projection, transposed)
//   MODE 2: f32 out, plain [M,N]               (final projection)
// ---------------------------------------------------------------------------
template <int MODE>
__global__ void __launch_bounds__(128)
gemm_nt(const unsigned short* __restrict__ A,
        const unsigned short* __restrict__ Bw,
        const float* __restrict__ bias,
        void* __restrict__ out,
        int M, int N, int K) {
    const int lane = threadIdx.x & 31;
    const int wid  = blockIdx.x * (blockDim.x >> 5) + (threadIdx.x >> 5);
    const int ntN  = N >> 6;
    const int tM = wid / ntN;
    const int tN = wid % ntN;
    if (tM * 64 >= M) return;
    const int m0 = tM * 64;
    const int n0 = tN * 64;

    v8f c[4][4];
    #pragma unroll
    for (int i = 0; i < 4; ++i)
        #pragma unroll
        for (int j = 0; j < 4; ++j)
            #pragma unroll
            for (int r = 0; r < 8; ++r) c[i][j][r] = 0.0f;

    for (int k0 = 0; k0 < K; k0 += 32) {
        if (k0 + 32 < K) {  // prefetch next K panel (one row per lane)
            __builtin_prefetch(A  + (size_t)(m0 + 2 * lane) * K + k0 + 32, 0, 3);
            __builtin_prefetch(Bw + (size_t)(n0 + 2 * lane) * K + k0 + 32, 0, 3);
        }
        FragBF16 a[4], b[4];
        #pragma unroll
        for (int mt = 0; mt < 4; ++mt)
            load_frag_a(a[mt], A + (size_t)(m0 + mt * 16) * K + k0, K, lane);
        #pragma unroll
        for (int nt = 0; nt < 4; ++nt)
            load_frag_b(b[nt], Bw + (size_t)(n0 + nt * 16) * K + k0, K, lane);
        #pragma unroll
        for (int mt = 0; mt < 4; ++mt)
            #pragma unroll
            for (int nt = 0; nt < 4; ++nt)
                c[mt][nt] = wmma_bf16(a[mt], b[nt], c[mt][nt]);
    }

    const int colL   = lane & 15;
    const int rowoff = (lane >> 4) << 3;
    #pragma unroll
    for (int mt = 0; mt < 4; ++mt) {
        #pragma unroll
        for (int nt = 0; nt < 4; ++nt) {
            const int   n    = n0 + nt * 16 + colL;
            const float bval = bias[n];
            #pragma unroll
            for (int r = 0; r < 8; ++r) {
                const int   m   = m0 + mt * 16 + r + rowoff;
                const float val = c[mt][nt][r] + bval;
                if (MODE == 2) {
                    ((float*)out)[(size_t)m * N + n] = val;
                } else {
                    const int b_  = m / TT;
                    const int t_  = m % TT;
                    const int h_  = n / HD;
                    const int hd_ = n % HD;
                    size_t dst;
                    if (MODE == 0)
                        dst = (((size_t)(b_ * HH + h_) * TT + t_) * HD + hd_);
                    else
                        dst = (((size_t)(b_ * HH + h_) * HD + hd_) * TT + t_);
                    ((unsigned short*)out)[dst] = f2bf(val);
                }
            }
        }
    }
}

// ---------------------------------------------------------------------------
// Kernel 3: flash attention. One wave (block=32) per (b*h, 16-row q tile).
//   Q,K: [BH, T, HD] bf16    Vt: [BH, HD, T] bf16
//   out: attn_out [B*T, D] bf16 (row-major, heads re-interleaved)
//
// Softmax strategy: S tile (C-layout) -> LDS (f32) -> re-read in A-fragment
// layout, so each lane owns half a row; row reductions are 15 in-register
// ops + ONE shfl_xor(16). P is packed into the A-fragment in registers
// (no second LDS trip). Only alpha / 1/l hop back to C-layout (8 gathers).
// launch_bounds(32) + waves_per_eu(1) keep everything resident (no spills).
// All four K B-fragments are loaded before the first WMMA so the waits
// stage (loadcnt<=2, <=0) instead of serializing load->wait(0)->wmma.
// ---------------------------------------------------------------------------
__global__ void __launch_bounds__(32)
__attribute__((amdgpu_waves_per_eu(1)))
attn_flash(const unsigned short* __restrict__ qb,
           const unsigned short* __restrict__ kbuf,
           const unsigned short* __restrict__ vtb,
           unsigned short* __restrict__ aob) {
    __shared__ __align__(16) float lsS[16 * 32];  // raw S tile, 2 KB

    const int lane = threadIdx.x;
    const int bh   = blockIdx.x;           // 0..B*H-1
    const int qt   = blockIdx.y;           // 0..T/16-1
    const int q0   = qt * 16;

    const unsigned short* Q  = qb   + (size_t)bh * TT * HD;
    const unsigned short* Km = kbuf + (size_t)bh * TT * HD;
    const unsigned short* Vt = vtb  + (size_t)bh * HD * TT;

    FragBF16 aq[2];
    load_frag_a(aq[0], Q + (size_t)q0 * HD,      HD, lane);
    load_frag_a(aq[1], Q + (size_t)q0 * HD + 32, HD, lane);

    float mrun = -3.0e38f;   // running max of this lane's row (row = lane&15)
    float lrun = 0.0f;       // running sum of this lane's row
    v8f o[4];
    #pragma unroll
    for (int d = 0; d < 4; ++d)
        #pragma unroll
        for (int r = 0; r < 8; ++r) o[d][r] = 0.0f;

    const int colL   = lane & 15;
    const int rowoff = (lane >> 4) << 3;
    const int arow   = lane & 15;          // A-layout row this lane owns
    const int akb    = (lane >> 4) * 8;    // A-layout K base

    for (int k0 = 0; k0 < TT; k0 += 32) {
        if (k0 + 32 < TT) {
            __builtin_prefetch(Km + (size_t)(k0 + 32 + colL) * HD, 0, 3);
            __builtin_prefetch(Vt + (size_t)(lane * 2) * TT + k0 + 32, 0, 3);
        }

        // ---- batch ALL K B-fragment loads, then the 4 WMMAs ----
        FragBF16 bk[2][2];
        #pragma unroll
        for (int j = 0; j < 2; ++j) {
            load_frag_b(bk[j][0], Km + (size_t)(k0 + j * 16) * HD,      HD, lane);
            load_frag_b(bk[j][1], Km + (size_t)(k0 + j * 16) * HD + 32, HD, lane);
        }
        v8f s[2];
        #pragma unroll
        for (int j = 0; j < 2; ++j) {
            v8f z;
            #pragma unroll
            for (int r = 0; r < 8; ++r) z[r] = 0.0f;
            z = wmma_bf16(aq[0], bk[j][0], z);
            s[j] = wmma_bf16(aq[1], bk[j][1], z);
        }
        #pragma unroll
        for (int j = 0; j < 2; ++j)
            #pragma unroll
            for (int r = 0; r < 8; ++r)
                lsS[(r + rowoff) * 32 + j * 16 + colL] = s[j][r] * 0.125f;
        __syncthreads();

        // ---- re-read own row half in A-fragment order ----
        float p16[16];
        {
            const float* rp = lsS + arow * 32;
            v4f c0 = *reinterpret_cast<const v4f*>(rp + akb);
            v4f c1 = *reinterpret_cast<const v4f*>(rp + akb + 4);
            v4f c2 = *reinterpret_cast<const v4f*>(rp + akb + 16);
            v4f c3 = *reinterpret_cast<const v4f*>(rp + akb + 20);
            #pragma unroll
            for (int i = 0; i < 4; ++i) {
                p16[i]      = c0[i];
                p16[4 + i]  = c1[i];
                p16[8 + i]  = c2[i];
                p16[12 + i] = c3[i];
            }
        }
        __syncthreads();

        // ---- online softmax: local reduce + one cross-half exchange ----
        float mloc = p16[0];
        #pragma unroll
        for (int i = 1; i < 16; ++i) mloc = fmaxf(mloc, p16[i]);
        mloc = fmaxf(mloc, __shfl_xor(mloc, 16, 32));
        const float mnew  = fmaxf(mrun, mloc);
        const float alpha = __expf(mrun - mnew);
        mrun = mnew;

        float ssum = 0.0f;
        #pragma unroll
        for (int i = 0; i < 16; ++i) {
            p16[i] = __expf(p16[i] - mnew);
            ssum += p16[i];
        }
        ssum += __shfl_xor(ssum, 16, 32);
        lrun = lrun * alpha + ssum;

        // ---- pack P directly into A-fragment registers ----
        FragBF16 pf;
        #pragma unroll
        for (int i = 0; i < 8; ++i)
            pf.u[i] = pack_bf16(p16[2 * i], p16[2 * i + 1]);

        // ---- gather per-row alpha into C-layout, rescale O ----
        float aC[8];
        #pragma unroll
        for (int r = 0; r < 8; ++r) aC[r] = __shfl(alpha, r + rowoff, 32);
        #pragma unroll
        for (int d = 0; d < 4; ++d)
            #pragma unroll
            for (int r = 0; r < 8; ++r) o[d][r] *= aC[r];

        // ---- O += P @ V^T  (Vt rows are contiguous along k) ----
        FragBF16 vf[4];
        #pragma unroll
        for (int d = 0; d < 4; ++d)
            load_frag_b(vf[d], Vt + (size_t)(d * 16) * TT + k0, TT, lane);
        #pragma unroll
        for (int d = 0; d < 4; ++d)
            o[d] = wmma_bf16(pf, vf[d], o[d]);
    }

    // ---- finalize: divide by row sums, scatter to [B*T, D] bf16 ----
    const float inv = 1.0f / lrun;
    float invC[8];
    #pragma unroll
    for (int r = 0; r < 8; ++r) invC[r] = __shfl(inv, r + rowoff, 32);

    const int b_ = bh / HH;
    const int h_ = bh % HH;
    #pragma unroll
    for (int d = 0; d < 4; ++d) {
        #pragma unroll
        for (int r = 0; r < 8; ++r) {
            const int t_ = q0 + r + rowoff;
            const int e_ = h_ * HD + d * 16 + colL;
            aob[(size_t)(b_ * TT + t_) * DD + e_] = f2bf(o[d][r] * invC[r]);
        }
    }
}

// ---------------------------------------------------------------------------
// Host-side orchestration
// ---------------------------------------------------------------------------
extern "C" void kernel_launch(void* const* d_in, const int* in_sizes, int n_in,
                              void* d_out, int out_size, void* d_ws, size_t ws_size,
                              hipStream_t stream) {
    const float* x  = (const float*)d_in[0];
    const float* Wq = (const float*)d_in[1];
    const float* bq = (const float*)d_in[2];
    const float* Wk = (const float*)d_in[3];
    const float* bk = (const float*)d_in[4];
    const float* Wv = (const float*)d_in[5];
    const float* bv = (const float*)d_in[6];
    const float* Wo = (const float*)d_in[7];
    const float* bo = (const float*)d_in[8];
    float* out = (float*)d_out;

    const size_t MB = 1024ull * 1024ull;
    char* ws = (char*)d_ws;
    unsigned short* xb  = (unsigned short*)(ws + 0);        // 16 MB, x bf16 [8192,1024]
    unsigned short* wqb = (unsigned short*)(ws + 16 * MB);  // 2 MB each
    unsigned short* wkb = (unsigned short*)(ws + 18 * MB);
    unsigned short* wvb = (unsigned short*)(ws + 20 * MB);
    unsigned short* wob = (unsigned short*)(ws + 22 * MB);
    unsigned short* qb  = (unsigned short*)(ws + 24 * MB);  // [B,H,T,HD] 16 MB
    unsigned short* kb  = (unsigned short*)(ws + 40 * MB);  // [B,H,T,HD] 16 MB
    unsigned short* vtb = (unsigned short*)(ws + 56 * MB);  // [B,H,HD,T] 16 MB
    unsigned short* aob = xb;  // attn out [B*T, D]; x bf16 is dead by then

    const int M = BB * TT;      // 8192
    const int N = DD;           // 1024
    const int K = DD;           // 1024

    // 1) fp32 -> bf16 (4 elems/thread)
    cvt_f32_bf16x4<<<(M * K / 4 + 255) / 256, 256, 0, stream>>>(x, xb, M * K / 4);
    cvt_f32_bf16x4<<<(DD * DD / 4 + 255) / 256, 256, 0, stream>>>(Wq, wqb, DD * DD / 4);
    cvt_f32_bf16x4<<<(DD * DD / 4 + 255) / 256, 256, 0, stream>>>(Wk, wkb, DD * DD / 4);
    cvt_f32_bf16x4<<<(DD * DD / 4 + 255) / 256, 256, 0, stream>>>(Wv, wvb, DD * DD / 4);
    cvt_f32_bf16x4<<<(DD * DD / 4 + 255) / 256, 256, 0, stream>>>(Wo, wob, DD * DD / 4);

    // 2) Q/K/V projections (WMMA NT GEMMs), 64x64 tile per wave, 4 waves/block
    const int tiles  = (M / 64) * (N / 64);   // 2048
    const int blocks = tiles / 4;             // 512
    gemm_nt<0><<<blocks, 128, 0, stream>>>(xb, wqb, bq, qb,  M, N, K);
    gemm_nt<0><<<blocks, 128, 0, stream>>>(xb, wkb, bk, kb,  M, N, K);
    gemm_nt<1><<<blocks, 128, 0, stream>>>(xb, wvb, bv, vtb, M, N, K);

    // 3) flash attention: one wave per (b*h, q-tile)
    attn_flash<<<dim3(BB * HH, TT / 16), 32, 0, stream>>>(qb, kb, vtb, aob);

    // 4) output projection, fp32 result
    gemm_nt<2><<<blocks, 128, 0, stream>>>(aob, wob, bo, out, M, N, K);
}
